// DeepseekV3TopkRouter_86526411145671
// MI455X (gfx1250) — matrix operand order repaired and checked by reading, not compiled, not run on hardware
//
#include <hip/hip_runtime.h>
#include <stdint.h>

// ---------------- constants ----------------
#define NEXP   256     // routed experts
#define HDIM   2048    // hidden size
#define TOPK   8
#define SCALING 2.5f

#define MTILE  32      // tokens per workgroup
#define KC     32      // K chunk staged in LDS (one WMMA k-step)
#define AROWB  144     // bytes per A row in LDS: 32 fp32 + pad (16B aligned, 36w stride conflict-free)
#define BROWB  80      // bytes per B row in LDS: 32 bf16 + pad (16B aligned, 20w stride conflict-free)
#define ABYTES (MTILE * AROWB)    // 4608
#define BBYTES (NEXP * BROWB)     // 20480
#define BUFB   (ABYTES + BBYTES)  // 25088 per buffer, x2 double-buffered
#define SPAD   257     // floats per score row (conflict-free token stride)

#define NEG_INF (-3.402823466e+38f)

typedef __attribute__((ext_vector_type(16))) __bf16 bf16x16;
typedef __attribute__((ext_vector_type(8)))  float  f32x8;

// fp32 -> bf16 RNE for the one-time weight prep
__device__ __forceinline__ unsigned short f2bf(float f) {
    unsigned x = __float_as_uint(f);
    x += 0x7fffu + ((x >> 16) & 1u);
    return (unsigned short)(x >> 16);
}

// CDNA5 async DMA: global -> LDS, 16B per lane, tracked by ASYNCcnt (ISA ch.10/§15.18)
__device__ __forceinline__ void async_b128(unsigned lds, unsigned goff, const void* base) {
    asm volatile("global_load_async_to_lds_b128 %0, %1, %2"
                 :: "v"(lds), "v"(goff), "s"(base) : "memory");
}
__device__ __forceinline__ void wait_async0() {
    asm volatile("s_wait_asynccnt 0" ::: "memory");
}

// ---------------- kernel 0: fp32 weight -> bf16 (2MB -> 1MB, one-time) ----------------
__global__ void prep_weights(const float* __restrict__ w,
                             unsigned short* __restrict__ wb, int n) {
    int i = blockIdx.x * blockDim.x + threadIdx.x;
    if (i < n) wb[i] = f2bf(w[i]);
}

// ---------------- kernel 1: fused GEMM (bf16 WMMA, async double-buffer) + top-k ----------------
__global__ __launch_bounds__(256)
void router_kernel(const float* __restrict__ x,            // [N, 2048] fp32
                   const unsigned short* __restrict__ wb,  // [256, 2048] bf16
                   const float* __restrict__ bias,         // [256] fp32
                   int* __restrict__ oIdx,                 // [N, 8]
                   float* __restrict__ oW,                 // [N, 8]
                   float* __restrict__ oLog)               // [N, 256]
{
    __shared__ __align__(16) unsigned char smem[2 * BUFB];   // 50176 B
    float* Sc  = (float*)smem;                          // routing phase: [32][257] f32
    float* Bls = (float*)(smem + MTILE * SPAD * 4);     // routing phase: [256] bias

    const int tid   = threadIdx.x;
    const int wave  = tid >> 5;
    const int lane  = tid & 31;
    const int mBase = blockIdx.x * MTILE;

    const int mhalf = wave & 1;          // which 16-token half
    const int eBase = (wave >> 1) * 64;  // 64-expert slice per wave pair
    const int l16   = lane & 15;
    const int halfA = (lane < 16) ? 0 : 8;   // accumulator row offset (lanes 16-31 hold M+8)
    const int halfB = (lane < 16) ? 0 : 16;  // A/B-frag K base (contiguous k = 16h + s)
    const int arow  = mhalf * 16 + l16;

    const unsigned ldsBase = (unsigned)(uintptr_t)(void*)smem;  // low 32b = LDS offset

    // per-thread staging slots (constant across chunks)
    const int aRow = tid >> 3, aC = tid & 7;          // A: 32 rows x 8 x 16B
    const unsigned aGRow = (unsigned)((mBase + aRow) * (HDIM * 4));

    f32x8 acc[4] = {};   // 4 N-tiles of 16 experts each

    // issue chunk 0 into buffer 0
    {
        unsigned ab = ldsBase;
        async_b128(ab + aRow * AROWB + aC * 16, aGRow + aC * 16, x);
        unsigned bb = ab + ABYTES;
#pragma unroll
        for (int j = 0; j < 4; ++j) {
            int s = tid + j * 256;                    // B: 256 rows x 4 x 16B
            int row = s >> 2, c = s & 3;
            async_b128(bb + row * BROWB + c * 16,
                       (unsigned)(row * (HDIM * 2) + c * 16), wb);
        }
    }

    const int nchunks = HDIM / KC;   // 64
    for (int ck = 0; ck < nchunks; ++ck) {
        const int buf = ck & 1;
        wait_async0();        // my wave's async copies into `buf` have landed
        __syncthreads();      // everyone's have

        if (ck + 1 < nchunks) {          // prefetch next chunk into other buffer
            const int kb = (ck + 1) * KC;
            unsigned ab = ldsBase + (buf ^ 1) * BUFB;
            async_b128(ab + aRow * AROWB + aC * 16, aGRow + kb * 4 + aC * 16, x);
            unsigned bb = ab + ABYTES;
#pragma unroll
            for (int j = 0; j < 4; ++j) {
                int s = tid + j * 256;
                int row = s >> 2, c = s & 3;
                async_b128(bb + row * BROWB + c * 16,
                           (unsigned)(row * (HDIM * 2) + kb * 2 + c * 16), wb);
            }
        }

        // ---- compute: one 16x16x32 WMMA k-step over 4 expert tiles ----
        const float*    Ald = (const float*)(smem + buf * BUFB);
        const uint32_t* Bld = (const uint32_t*)(smem + buf * BUFB + ABYTES);

        union { bf16x16 v; uint32_t u[8]; } af;
        const float* ap = Ald + arow * (AROWB / 4);
#pragma unroll
        for (int j = 0; j < 8; ++j) {
            int k = halfB + 2 * j;
            af.v[2 * j]     = (__bf16)ap[k];        // native v_cvt bf16 path
            af.v[2 * j + 1] = (__bf16)ap[k + 1];
        }
#pragma unroll
        for (int nt = 0; nt < 4; ++nt) {
            int e = eBase + nt * 16 + l16;
            const uint32_t* bp = Bld + e * (BROWB / 4) + (halfB >> 1);
            union { bf16x16 v; uint32_t u[8]; } bf;
#pragma unroll
            for (int j = 0; j < 8; ++j) bf.u[j] = bp[j];
            acc[nt] = __builtin_amdgcn_wmma_f32_16x16x32_bf16(
                false, af.v, false, bf.v, (short)0, acc[nt], false, false);
        }
    }
    __syncthreads();  // staging buffers dead; safe to alias with scores

    // ---- epilogue: logits to global, sigmoid scores + bias to LDS ----
    Bls[tid] = bias[tid];  // tid < 256 == NEXP
#pragma unroll
    for (int nt = 0; nt < 4; ++nt) {
        int e = eBase + nt * 16 + l16;
#pragma unroll
        for (int v = 0; v < 8; ++v) {
            int tl = mhalf * 16 + v + halfA;  // C-layout: lanes 16-31 hold M+8
            float val = acc[nt][v];
            oLog[(size_t)(mBase + tl) * NEXP + e] = val;
            Sc[tl * SPAD + e] = 1.0f / (1.0f + __expf(-val));
        }
    }
    __syncthreads();

    // ---- routing: 8 lanes per token, 4 tokens per wave ----
    const int sub  = lane & 7;
    const int tLoc = wave * 4 + (lane >> 3);
    const int tok  = mBase + tLoc;
    const float* sr = Sc + tLoc * SPAD;

    // 1) group scores = sum of top-2 (scores+bias) per 32-expert group
    float gs[8];
#pragma unroll
    for (int g = 0; g < 8; ++g) {
        float m1 = NEG_INF, m2 = NEG_INF;
#pragma unroll
        for (int i = 0; i < 4; ++i) {
            int e = g * 32 + sub * 4 + i;
            float s = sr[e] + Bls[e];
            if (s > m1) { m2 = m1; m1 = s; }
            else if (s > m2) { m2 = s; }
        }
#pragma unroll
        for (int m = 1; m < 8; m <<= 1) {   // merge top-2 across 8 lanes
            float o1 = __shfl_xor(m1, m, 32);
            float o2 = __shfl_xor(m2, m, 32);
            if (o1 > m1) { m2 = fmaxf(m1, o2); m1 = o1; }
            else         { m2 = fmaxf(m2, o1); }
        }
        gs[g] = m1 + m2;
    }

    // 2) top-4 groups (strict > keeps lowest index on ties, matching lax.top_k)
    unsigned gmask = 0;
#pragma unroll
    for (int r = 0; r < 4; ++r) {
        float bv = NEG_INF; int bi = 0;
#pragma unroll
        for (int g = 0; g < 8; ++g) {
            if (gs[g] > bv) { bv = gs[g]; bi = g; }
        }
        gmask |= 1u << bi;
#pragma unroll
        for (int g = 0; g < 8; ++g) {
            if (g == bi) gs[g] = NEG_INF;
        }
    }

    // 3) lane-local top-8 over this lane's 32 experts (masked score = -1.0)
    float lv[8]; int li[8];
#pragma unroll
    for (int k = 0; k < 8; ++k) { lv[k] = NEG_INF; li[k] = 0x7fffffff; }
#pragma unroll
    for (int g = 0; g < 8; ++g) {
        bool sel = (gmask >> g) & 1u;
#pragma unroll
        for (int i = 0; i < 4; ++i) {
            int e = g * 32 + sub * 4 + i;
            float s = sel ? (sr[e] + Bls[e]) : -1.0f;
            if (s > lv[7]) {                   // static-index bubble insertion
                float cv = s; int ci = e;
#pragma unroll
                for (int k = 0; k < 8; ++k) {
                    bool gt = cv > lv[k];
                    float tv = gt ? lv[k] : cv; int ti = gt ? li[k] : ci;
                    lv[k] = gt ? cv : lv[k];    li[k] = gt ? ci : li[k];
                    cv = tv; ci = ti;
                }
            }
        }
    }

    // 4) merge 8 lane-local lists -> global top-8 (tie -> lower expert index)
    int myI = 0;
#pragma unroll
    for (int k = 0; k < 8; ++k) {
        float bv = lv[0]; int bi = li[0]; int bl = sub;
#pragma unroll
        for (int m = 1; m < 8; m <<= 1) {
            float ov = __shfl_xor(bv, m, 32);
            int   oi = __shfl_xor(bi, m, 32);
            int   ol = __shfl_xor(bl, m, 32);
            bool take = (ov > bv) || (ov == bv && oi < bi);
            bv = take ? ov : bv; bi = take ? oi : bi; bl = take ? ol : bl;
        }
        if (k == sub) myI = bi;       // lane `sub` owns output slot k
        if (bl == sub) {              // winner pops its head
#pragma unroll
            for (int k2 = 0; k2 < 7; ++k2) { lv[k2] = lv[k2 + 1]; li[k2] = li[k2 + 1]; }
            lv[7] = NEG_INF; li[7] = 0x7fffffff;
        }
    }

    // 5) gather weight from scores (no bias), normalize, scale
    float wsel = sr[myI];
    float s8 = wsel;
#pragma unroll
    for (int m = 1; m < 8; m <<= 1) s8 += __shfl_xor(s8, m, 32);
    float outw = wsel / (s8 + 1e-20f) * SCALING;

    oIdx[tok * TOPK + sub] = myI;
    oW[tok * TOPK + sub]   = outw;
}

// ---------------- launcher ----------------
extern "C" void kernel_launch(void* const* d_in, const int* in_sizes, int n_in,
                              void* d_out, int out_size, void* d_ws, size_t ws_size,
                              hipStream_t stream) {
    const float* x    = (const float*)d_in[0];   // [4,4096,2048] fp32
    const float* w    = (const float*)d_in[1];   // [256,2048] fp32
    const float* bias = (const float*)d_in[2];   // [256] fp32

    const int N = in_sizes[0] / HDIM;            // 16384 tokens

    // workspace: bf16 weights (256*2048*2 = 1 MB)
    unsigned short* wb = (unsigned short*)d_ws;
    const int nw = NEXP * HDIM;
    prep_weights<<<(nw + 255) / 256, 256, 0, stream>>>(w, wb, nw);

    float* out  = (float*)d_out;
    int*   oIdx = (int*)out;                       // [N,8] int32 (raw bits)
    float* oW   = out + (size_t)N * TOPK;          // [N,8] fp32
    float* oLog = out + (size_t)N * TOPK * 2;      // [N,256] fp32

    router_kernel<<<N / MTILE, 256, 0, stream>>>(x, wb, bias, oIdx, oW, oLog);
}
